// Dynamic_AgentAttn_5119601017203
// MI455X (gfx1250) — matrix operand order repaired and checked by reading, not compiled, not run on hardware
//
#include <hip/hip_runtime.h>
#include <hip/hip_bf16.h>

// ---- problem constants --------------------------------------------------
#define DIMC   512
#define NHEADS 8
#define AGENTN 86
#define AGP    96      // agent dim padded to 6 WMMA tiles
#define BATCH  16
#define WSZ    56
#define NTOK   3136    // 56*56
#define HDIM   64
#define SCL    0.125f  // 64^-0.5

typedef __attribute__((ext_vector_type(16))) __bf16 bf16x16;
typedef __attribute__((ext_vector_type(8)))  __bf16 bf16x8;
typedef __attribute__((ext_vector_type(2)))  __bf16 bf16x2;
typedef __attribute__((ext_vector_type(8)))  float  f32x8;
typedef __attribute__((ext_vector_type(4)))  int    v4i;

// ---- optional CDNA5 async global->LDS path ------------------------------
#ifdef __has_builtin
#if __has_builtin(__builtin_amdgcn_global_load_async_to_lds_b128)
#define USE_ASYNC_LDS 1
#endif
#endif

#ifdef USE_ASYNC_LDS
__device__ __forceinline__ void async_cp_b128(const void* g, void* l) {
  __builtin_amdgcn_global_load_async_to_lds_b128(
      (__attribute__((address_space(1))) v4i*)(g),
      (__attribute__((address_space(3))) v4i*)(l), 0, 0);
}
__device__ __forceinline__ void async_wait0() {
#if __has_builtin(__builtin_amdgcn_s_wait_asynccnt)
  __builtin_amdgcn_s_wait_asynccnt(0);
#else
  asm volatile("s_wait_asynccnt 0x0" ::: "memory");
#endif
}
#endif

// ---- WMMA helpers -------------------------------------------------------
__device__ __forceinline__ f32x8 wmma_bf16(bf16x16 a, bf16x16 b, f32x8 c) {
  return __builtin_amdgcn_wmma_f32_16x16x32_bf16(false, a, false, b,
                                                 (short)0, c, false, false);
}

// A-fragment (16x32 bf16, row-major source, row stride ld):
// lanes 0-15:  row m=lane,    K 0..7 / 16..23 ; lanes 16-31: K 8..15 / 24..31
__device__ __forceinline__ bf16x16 load_a_frag(const __bf16* p, int ld) {
  int lane = threadIdx.x & 31;
  int m    = lane & 15;
  int kb   = (lane & 16) ? 8 : 0;
  const __bf16* q = p + m * ld + kb;
  bf16x8 lo = *(const bf16x8*)(q);
  bf16x8 hi = *(const bf16x8*)(q + 16);
  bf16x16 r;
#pragma unroll
  for (int i = 0; i < 8; ++i) { r[i] = lo[i]; r[i + 8] = hi[i]; }
  return r;
}

// B-fragment (32x16 bf16). Source holds B transposed: (col n) x (K), stride ld.
__device__ __forceinline__ bf16x16 load_b_frag(const __bf16* p, int ld) {
  int lane = threadIdx.x & 31;
  int n    = lane & 15;
  int kb   = (lane & 16) ? 8 : 0;
  const __bf16* q = p + n * ld + kb;
  bf16x8 lo = *(const bf16x8*)(q);
  bf16x8 hi = *(const bf16x8*)(q + 16);
  bf16x16 r;
#pragma unroll
  for (int i = 0; i < 8; ++i) { r[i] = lo[i]; r[i + 8] = hi[i]; }
  return r;
}

// ---- kernel 1: pack weights / agent tokens to bf16 ----------------------
__global__ void cvt_pack(const float* __restrict__ q_w, const float* __restrict__ kv_w,
                         const float* __restrict__ proj_w, const float* __restrict__ agt,
                         __bf16* __restrict__ wqkv, __bf16* __restrict__ wproj,
                         __bf16* __restrict__ agbf) {
  int idx = blockIdx.x * blockDim.x + threadIdx.x;
  const int T1 = DIMC * 1536, T2 = DIMC * DIMC, T3 = NHEADS * AGP * HDIM;
  if (idx < T1) {
    int k = idx / 1536, j = idx % 1536;
    float v = (j < DIMC) ? q_w[k * DIMC + j] : kv_w[k * (2 * DIMC) + (j - DIMC)];
    wqkv[idx] = (__bf16)v;
  } else if (idx < T1 + T2) {
    int i = idx - T1;
    wproj[i] = (__bf16)proj_w[i];
  } else if (idx < T1 + T2 + T3) {
    int i = idx - T1 - T2;
    int h = i / (AGP * HDIM), r = i % (AGP * HDIM);
    int a = r / HDIM, d = r % HDIM;
    float v = (a < AGENTN) ? agt[a * DIMC + h * HDIM + d] : 0.f;
    agbf[i] = (__bf16)v;
  }
}

// ---- kernel 2: bilinear 7x7 -> 56x56 (half-pixel centers) ---------------
__global__ void resize_bias(const float* __restrict__ an, const float* __restrict__ na,
                            float* __restrict__ pb1, float* __restrict__ nb1) {
  int idx = blockIdx.x * blockDim.x + threadIdx.x;
  const int per = NHEADS * AGENTN * NTOK;
  if (idx >= 2 * per) return;
  int which = idx / per, rem = idx % per;
  int ha = rem / NTOK, yx = rem % NTOK;
  int y = yx / WSZ, x = yx % WSZ;
  float sy = (y + 0.5f) * 0.125f - 0.5f;
  float sx = (x + 0.5f) * 0.125f - 0.5f;
  int y0 = (int)floorf(sy), x0 = (int)floorf(sx);
  float wy = sy - y0, wx = sx - x0;
  int y0c = min(max(y0, 0), 6), y1c = min(max(y0 + 1, 0), 6);
  int x0c = min(max(x0, 0), 6), x1c = min(max(x0 + 1, 0), 6);
  const float* s = (which ? na : an) + ha * 49;
  float v = (1.f - wy) * ((1.f - wx) * s[y0c * 7 + x0c] + wx * s[y0c * 7 + x1c]) +
            wy * ((1.f - wx) * s[y1c * 7 + x0c] + wx * s[y1c * 7 + x1c]);
  (which ? nb1 : pb1)[ha * NTOK + yx] = v;
}

// ---- kernel 3: QKV GEMM  (B*N,512) x (512,1536), A read transposed ------
// BM=64 BN=128 BK=64, 8 waves; 8 WMMA per wave per K-step.
__global__ __launch_bounds__(256) void gemm_qkv(
    const float* __restrict__ x, const __bf16* __restrict__ wqkv,
    __bf16* __restrict__ qb, __bf16* __restrict__ kbf, __bf16* __restrict__ vbf) {
  __shared__ __attribute__((aligned(16))) __bf16 As[64 * 64];   // (m,k)
  __shared__ __attribute__((aligned(16))) __bf16 Bs[128 * 64];  // (n,k)
  int t = threadIdx.x;
  int lane = t & 31, half = (lane >> 4) & 1, lcol = lane & 15;
  int wid = t >> 5, wm = wid >> 1, wn = wid & 1;
  int row0 = blockIdx.y * 64;
  int b = row0 / NTOK, n0 = row0 % NTOK;  // 64 | 3136 so no b-crossing
  int col0 = blockIdx.x * 128;
  f32x8 acc[4] = {};
  int kp = t & 31;          // k-pair id: k = 2*kp, 2*kp+1
  int mg = t >> 5;          // 0..7
  int m0 = mg * 8;
  int nb0 = (t >> 5) * 16;  // 0..112
  const float* xb = x + (size_t)b * DIMC * NTOK + n0;
  for (int k0 = 0; k0 < DIMC; k0 += 64) {
    // A tile: As[m][k] = x[b, k0+k, n0+m]; pack (k even, k odd) pairs.
    const float* a0 = xb + (size_t)(k0 + 2 * kp) * NTOK + m0;
    const float* a1 = a0 + NTOK;
    float4 e0 = *(const float4*)a0;
    float4 e1 = *(const float4*)(a0 + 4);
    float4 o0 = *(const float4*)a1;
    float4 o1 = *(const float4*)(a1 + 4);
    float ev[8] = {e0.x, e0.y, e0.z, e0.w, e1.x, e1.y, e1.z, e1.w};
    float ov[8] = {o0.x, o0.y, o0.z, o0.w, o1.x, o1.y, o1.z, o1.w};
#pragma unroll
    for (int j = 0; j < 8; ++j) {
      bf16x2 pr; pr[0] = (__bf16)ev[j]; pr[1] = (__bf16)ov[j];
      *(bf16x2*)(As + (m0 + j) * 64 + 2 * kp) = pr;
    }
    // B tile transposed into (n,k); pack k pairs -> b32 stores.
    const __bf16* bp0 = wqkv + (size_t)(k0 + 2 * kp) * 1536 + col0 + nb0;
    const __bf16* bp1 = bp0 + 1536;
    bf16x8 q0 = *(const bf16x8*)bp0;
    bf16x8 q1 = *(const bf16x8*)(bp0 + 8);
    bf16x8 r0 = *(const bf16x8*)bp1;
    bf16x8 r1 = *(const bf16x8*)(bp1 + 8);
#pragma unroll
    for (int i = 0; i < 8; ++i) {
      bf16x2 pa; pa[0] = q0[i]; pa[1] = r0[i];
      *(bf16x2*)(Bs + (nb0 + i) * 64 + 2 * kp) = pa;
      bf16x2 pb; pb[0] = q1[i]; pb[1] = r1[i];
      *(bf16x2*)(Bs + (nb0 + 8 + i) * 64 + 2 * kp) = pb;
    }
    if (k0 + 64 < DIMC) {
      __builtin_prefetch(xb + (size_t)(k0 + 64 + 2 * kp) * NTOK + m0, 0, 1);
      __builtin_prefetch(wqkv + (size_t)(k0 + 64 + 2 * kp) * 1536 + col0 + nb0, 0, 1);
    }
    __syncthreads();
    bf16x16 af0 = load_a_frag(As + wm * 16 * 64, 64);
    bf16x16 af1 = load_a_frag(As + wm * 16 * 64 + 32, 64);
#pragma unroll
    for (int jn = 0; jn < 4; ++jn) {
      const __bf16* bsb = Bs + (wn * 64 + jn * 16) * 64;
      acc[jn] = wmma_bf16(af0, load_b_frag(bsb, 64), acc[jn]);
      acc[jn] = wmma_bf16(af1, load_b_frag(bsb + 32, 64), acc[jn]);
    }
    __syncthreads();
  }
#pragma unroll
  for (int jn = 0; jn < 4; ++jn) {
#pragma unroll
    for (int r = 0; r < 8; ++r) {
      int row = wm * 16 + r + 8 * half;
      int n = n0 + row;
      int j = col0 + wn * 64 + jn * 16 + lcol;
      __bf16 bv = (__bf16)acc[jn][r];
      if (j < DIMC) {
        qb[(((size_t)b * NHEADS + (j >> 6)) * NTOK + n) * HDIM + (j & 63)] = bv;
      } else if (j < 2 * DIMC) {
        int jj = j - DIMC;
        kbf[(((size_t)b * NHEADS + (jj >> 6)) * NTOK + n) * HDIM + (jj & 63)] = bv;
      } else {
        int jj = j - 2 * DIMC;
        vbf[(((size_t)b * NHEADS + (jj >> 6)) * NTOK + n) * HDIM + (jj & 63)] = bv;
      }
    }
  }
}

// ---- kernel 4: agent attention (flash-style over N) ---------------------
__global__ __launch_bounds__(192) void agent_attn(
    const __bf16* __restrict__ agbf, const __bf16* __restrict__ kbf,
    const __bf16* __restrict__ vbf, const float* __restrict__ pb1,
    __bf16* __restrict__ agv) {
  __shared__ __attribute__((aligned(16))) __bf16 k_lds[64 * 64];   // (n,d)
  __shared__ __attribute__((aligned(16))) __bf16 v_lds[64 * 64];   // (d,n)
  __shared__ __attribute__((aligned(16))) __bf16 pst[6 * 16 * 64]; // P staging
  int t = threadIdx.x;
  int lane = t & 31, half = (lane >> 4) & 1, lcol = lane & 15;
  int wid = t >> 5;                // 6 waves, one 16-agent-row tile each
  int bh = blockIdx.x, h = bh & 7;
  const __bf16* kg = kbf + (size_t)bh * NTOK * HDIM;
  const __bf16* vg = vbf + (size_t)bh * NTOK * HDIM;
  const __bf16* agh = agbf + (size_t)h * AGP * HDIM + wid * 16 * HDIM;
  bf16x16 afr0 = load_a_frag(agh, HDIM);
  bf16x16 afr1 = load_a_frag(agh + 32, HDIM);
  float rmax[8], rsum[8];
  f32x8 O[4] = {};
#pragma unroll
  for (int r = 0; r < 8; ++r) { rmax[r] = -1e30f; rsum[r] = 0.f; }
  __bf16* mypst = pst + wid * 16 * 64;
  for (int nc = 0; nc < NTOK / 64; ++nc) {
    int n0 = nc * 64;
#ifdef USE_ASYNC_LDS
    for (int i = t; i < 64 * 8; i += 192) {            // K chunk: async bulk copy
      int n = i >> 3, dv = (i & 7) * 8;
      async_cp_b128(kg + (size_t)(n0 + n) * HDIM + dv, k_lds + n * 64 + dv);
    }
#else
    for (int i = t; i < 64 * 8; i += 192) {
      int n = i >> 3, dv = (i & 7) * 8;
      *(bf16x8*)(k_lds + n * 64 + dv) =
          *(const bf16x8*)(kg + (size_t)(n0 + n) * HDIM + dv);
    }
#endif
    for (int u = t; u < 256; u += 192) {               // V chunk, transposed, packed
      int np = (u >> 3) * 2;
      int dv = (u & 7) * 8;
      bf16x8 r0 = *(const bf16x8*)(vg + (size_t)(n0 + np) * HDIM + dv);
      bf16x8 r1 = *(const bf16x8*)(vg + (size_t)(n0 + np + 1) * HDIM + dv);
#pragma unroll
      for (int i = 0; i < 8; ++i) {
        bf16x2 pr; pr[0] = r0[i]; pr[1] = r1[i];
        *(bf16x2*)(v_lds + (dv + i) * 64 + np) = pr;
      }
    }
#ifdef USE_ASYNC_LDS
    async_wait0();
#endif
    __syncthreads();
    f32x8 S[4];
#pragma unroll
    for (int jn = 0; jn < 4; ++jn) {
      f32x8 s = {};
      s = wmma_bf16(afr0, load_b_frag(k_lds + jn * 16 * 64, 64), s);
      s = wmma_bf16(afr1, load_b_frag(k_lds + jn * 16 * 64 + 32, 64), s);
#pragma unroll
      for (int r = 0; r < 8; ++r) {
        int a = wid * 16 + r + 8 * half;
        int n = n0 + jn * 16 + lcol;
        float bias = (a < AGENTN) ? pb1[((size_t)h * AGENTN + a) * NTOK + n] : 0.f;
        s[r] = s[r] * SCL + bias;
      }
      S[jn] = s;
    }
    float nm[8], alpha[8];
#pragma unroll
    for (int r = 0; r < 8; ++r) {
      float m = fmaxf(fmaxf(S[0][r], S[1][r]), fmaxf(S[2][r], S[3][r]));
#pragma unroll
      for (int sh = 1; sh < 16; sh <<= 1) m = fmaxf(m, __shfl_xor(m, sh, 32));
      nm[r] = fmaxf(rmax[r], m);
      alpha[r] = __expf(rmax[r] - nm[r]);
      rmax[r] = nm[r];
    }
#pragma unroll
    for (int jn = 0; jn < 4; ++jn)
#pragma unroll
      for (int r = 0; r < 8; ++r) S[jn][r] = __expf(S[jn][r] - nm[r]);
#pragma unroll
    for (int r = 0; r < 8; ++r) {
      float s = S[0][r] + S[1][r] + S[2][r] + S[3][r];
#pragma unroll
      for (int sh = 1; sh < 16; sh <<= 1) s += __shfl_xor(s, sh, 32);
      rsum[r] = rsum[r] * alpha[r] + s;
    }
#pragma unroll
    for (int jd = 0; jd < 4; ++jd)
#pragma unroll
      for (int r = 0; r < 8; ++r) O[jd][r] *= alpha[r];
#pragma unroll
    for (int jn = 0; jn < 4; ++jn)
#pragma unroll
      for (int r = 0; r < 8; ++r)
        mypst[(r + 8 * half) * 64 + jn * 16 + lcol] = (__bf16)S[jn][r];
    __syncthreads();
#pragma unroll
    for (int jd = 0; jd < 4; ++jd)
#pragma unroll
      for (int ks = 0; ks < 2; ++ks) {
        bf16x16 pa = load_a_frag(mypst + ks * 32, 64);
        bf16x16 vb = load_b_frag(v_lds + jd * 16 * 64 + ks * 32, 64);
        O[jd] = wmma_bf16(pa, vb, O[jd]);
      }
    __syncthreads();
  }
#pragma unroll
  for (int jd = 0; jd < 4; ++jd)
#pragma unroll
    for (int r = 0; r < 8; ++r) {
      int a = wid * 16 + r + 8 * half;
      int d = jd * 16 + lcol;
      agv[((size_t)bh * AGP + a) * HDIM + d] = (__bf16)(O[jd][r] / rsum[r]);
    }
}

// ---- kernel 5: query->agent attention + output --------------------------
__global__ __launch_bounds__(128) void q_attn(
    const __bf16* __restrict__ qbf, const __bf16* __restrict__ agbf,
    const __bf16* __restrict__ agv, const float* __restrict__ nb1,
    float* __restrict__ attn) {
  __shared__ __attribute__((aligned(16))) __bf16 ag_lds[AGP * HDIM];   // (a,d)
  __shared__ __attribute__((aligned(16))) __bf16 av_lds[HDIM * AGP];   // (d,a)
  __shared__ __attribute__((aligned(16))) __bf16 pst[4 * 16 * AGP];
  int t = threadIdx.x;
  int lane = t & 31, half = (lane >> 4) & 1, lcol = lane & 15;
  int wid = t >> 5;
  int bh = blockIdx.y, h = bh & 7, b = bh >> 3;
  int n0 = blockIdx.x * 64 + wid * 16;
#ifdef USE_ASYNC_LDS
  for (int i = t; i < AGP * HDIM / 8; i += 128) {
    int a = i >> 3, dv = (i & 7) * 8;
    async_cp_b128(agbf + (size_t)h * AGP * HDIM + a * HDIM + dv,
                  ag_lds + a * HDIM + dv);
  }
#else
  for (int i = t; i < AGP * HDIM / 8; i += 128) {
    int a = i >> 3, dv = (i & 7) * 8;
    *(bf16x8*)(ag_lds + a * HDIM + dv) =
        *(const bf16x8*)(agbf + (size_t)h * AGP * HDIM + a * HDIM + dv);
  }
#endif
  for (int u = t; u < 384; u += 128) {     // agent_v transposed, packed pairs
    int ap = (u >> 3) * 2;
    int dv = (u & 7) * 8;
    bf16x8 r0 = *(const bf16x8*)(agv + ((size_t)bh * AGP + ap) * HDIM + dv);
    bf16x8 r1 = *(const bf16x8*)(agv + ((size_t)bh * AGP + ap + 1) * HDIM + dv);
#pragma unroll
    for (int i = 0; i < 8; ++i) {
      bf16x2 pr; pr[0] = r0[i]; pr[1] = r1[i];
      *(bf16x2*)(av_lds + (dv + i) * AGP + ap) = pr;
    }
  }
#ifdef USE_ASYNC_LDS
  async_wait0();
#endif
  __syncthreads();
  const __bf16* qg = qbf + ((size_t)bh * NTOK + n0) * HDIM;
  bf16x16 qa0 = load_a_frag(qg, HDIM);
  bf16x16 qa1 = load_a_frag(qg + 32, HDIM);
  f32x8 S[6];
#pragma unroll
  for (int jn = 0; jn < 6; ++jn) {
    f32x8 s = {};
    s = wmma_bf16(qa0, load_b_frag(ag_lds + jn * 16 * HDIM, HDIM), s);
    s = wmma_bf16(qa1, load_b_frag(ag_lds + jn * 16 * HDIM + 32, HDIM), s);
#pragma unroll
    for (int r = 0; r < 8; ++r) {
      int a = jn * 16 + lcol;
      int n = n0 + r + 8 * half;
      s[r] = (a < AGENTN) ? (s[r] * SCL + nb1[((size_t)h * AGENTN + a) * NTOK + n])
                          : -1e30f;
    }
    S[jn] = s;
  }
  float rsum[8];
#pragma unroll
  for (int r = 0; r < 8; ++r) {
    float m = S[0][r];
#pragma unroll
    for (int jn = 1; jn < 6; ++jn) m = fmaxf(m, S[jn][r]);
#pragma unroll
    for (int sh = 1; sh < 16; sh <<= 1) m = fmaxf(m, __shfl_xor(m, sh, 32));
    float s = 0.f;
#pragma unroll
    for (int jn = 0; jn < 6; ++jn) { S[jn][r] = __expf(S[jn][r] - m); s += S[jn][r]; }
#pragma unroll
    for (int sh = 1; sh < 16; sh <<= 1) s += __shfl_xor(s, sh, 32);
    rsum[r] = s;
  }
  __bf16* mypst = pst + wid * 16 * AGP;
#pragma unroll
  for (int jn = 0; jn < 6; ++jn)
#pragma unroll
    for (int r = 0; r < 8; ++r)
      mypst[(r + 8 * half) * AGP + jn * 16 + lcol] = (__bf16)S[jn][r];
  __syncthreads();
  f32x8 O[4] = {};
#pragma unroll
  for (int jd = 0; jd < 4; ++jd)
#pragma unroll
    for (int ks = 0; ks < 3; ++ks) {
      bf16x16 pa = load_a_frag(mypst + ks * 32, AGP);
      bf16x16 vb = load_b_frag(av_lds + jd * 16 * AGP + ks * 32, AGP);
      O[jd] = wmma_bf16(pa, vb, O[jd]);
    }
#pragma unroll
  for (int jd = 0; jd < 4; ++jd)
#pragma unroll
    for (int r = 0; r < 8; ++r) {
      int n = n0 + r + 8 * half;
      int c = h * HDIM + jd * 16 + lcol;
      attn[((size_t)b * NTOK + n) * DIMC + c] = O[jd][r] / rsum[r];
    }
}

// ---- kernel 6: depthwise 3x3 conv on v, added into attn -----------------
__global__ void dwc_add(const __bf16* __restrict__ vbf, const float* __restrict__ dwc_w,
                        const float* __restrict__ dwc_b, float* __restrict__ attn) {
  long idx = (long)blockIdx.x * blockDim.x + threadIdx.x;
  const long total = (long)BATCH * DIMC * NTOK;
  if (idx >= total) return;
  int x = (int)(idx % WSZ); long r1 = idx / WSZ;
  int y = (int)(r1 % WSZ); long r2 = r1 / WSZ;
  int c = (int)(r2 % DIMC); int b = (int)(r2 / DIMC);
  float acc = dwc_b[c];
  const __bf16* vb = vbf + (((size_t)b * NHEADS + (c >> 6)) * NTOK) * HDIM + (c & 63);
#pragma unroll
  for (int dy = -1; dy <= 1; ++dy) {
    int yy = y + dy;
    if (yy < 0 || yy >= WSZ) continue;
#pragma unroll
    for (int dx = -1; dx <= 1; ++dx) {
      int xx = x + dx;
      if (xx < 0 || xx >= WSZ) continue;
      float w = dwc_w[c * 9 + (dy + 1) * 3 + (dx + 1)];
      acc += w * (float)vb[(size_t)(yy * WSZ + xx) * HDIM];
    }
  }
  attn[((size_t)b * NTOK + y * WSZ + x) * DIMC + c] += acc;
}

// ---- kernel 7: output projection GEMM + bias ----------------------------
__global__ __launch_bounds__(256) void gemm_proj(
    const float* __restrict__ attn, const __bf16* __restrict__ wproj,
    const float* __restrict__ proj_b, float* __restrict__ out) {
  __shared__ __attribute__((aligned(16))) __bf16 As[64 * 64];
  __shared__ __attribute__((aligned(16))) __bf16 Bs[128 * 64];
  int t = threadIdx.x;
  int lane = t & 31, half = (lane >> 4) & 1, lcol = lane & 15;
  int wid = t >> 5, wm = wid >> 1, wn = wid & 1;
  size_t row0 = (size_t)blockIdx.y * 64;
  int col0 = blockIdx.x * 128;
  f32x8 acc[4] = {};
  int mA = t >> 2, kA0 = (t & 3) * 16;   // A: 16 contiguous k per thread
  int kp = t & 31, nb0 = (t >> 5) * 16;  // B: k-pairs
  for (int k0 = 0; k0 < DIMC; k0 += 64) {
    const float* ap = attn + (row0 + mA) * DIMC + k0 + kA0;
    float4 f0 = *(const float4*)ap;
    float4 f1 = *(const float4*)(ap + 4);
    float4 f2 = *(const float4*)(ap + 8);
    float4 f3 = *(const float4*)(ap + 12);
    bf16x8 p0, p1;
    p0[0]=(__bf16)f0.x; p0[1]=(__bf16)f0.y; p0[2]=(__bf16)f0.z; p0[3]=(__bf16)f0.w;
    p0[4]=(__bf16)f1.x; p0[5]=(__bf16)f1.y; p0[6]=(__bf16)f1.z; p0[7]=(__bf16)f1.w;
    p1[0]=(__bf16)f2.x; p1[1]=(__bf16)f2.y; p1[2]=(__bf16)f2.z; p1[3]=(__bf16)f2.w;
    p1[4]=(__bf16)f3.x; p1[5]=(__bf16)f3.y; p1[6]=(__bf16)f3.z; p1[7]=(__bf16)f3.w;
    *(bf16x8*)(As + mA * 64 + kA0) = p0;       // contiguous b128 LDS stores
    *(bf16x8*)(As + mA * 64 + kA0 + 8) = p1;
    const __bf16* bp0 = wproj + (size_t)(k0 + 2 * kp) * DIMC + col0 + nb0;
    const __bf16* bp1 = bp0 + DIMC;
    bf16x8 q0 = *(const bf16x8*)bp0;
    bf16x8 q1 = *(const bf16x8*)(bp0 + 8);
    bf16x8 r0 = *(const bf16x8*)bp1;
    bf16x8 r1 = *(const bf16x8*)(bp1 + 8);
#pragma unroll
    for (int i = 0; i < 8; ++i) {
      bf16x2 pa; pa[0] = q0[i]; pa[1] = r0[i];
      *(bf16x2*)(Bs + (nb0 + i) * 64 + 2 * kp) = pa;
      bf16x2 pb; pb[0] = q1[i]; pb[1] = r1[i];
      *(bf16x2*)(Bs + (nb0 + 8 + i) * 64 + 2 * kp) = pb;
    }
    if (k0 + 64 < DIMC) {
      __builtin_prefetch(attn + (row0 + mA) * DIMC + k0 + 64 + kA0, 0, 1);
      __builtin_prefetch(wproj + (size_t)(k0 + 64 + 2 * kp) * DIMC + col0 + nb0, 0, 1);
    }
    __syncthreads();
    bf16x16 af0 = load_a_frag(As + wm * 16 * 64, 64);
    bf16x16 af1 = load_a_frag(As + wm * 16 * 64 + 32, 64);
#pragma unroll
    for (int jn = 0; jn < 4; ++jn) {
      const __bf16* bsb = Bs + (wn * 64 + jn * 16) * 64;
      acc[jn] = wmma_bf16(af0, load_b_frag(bsb, 64), acc[jn]);
      acc[jn] = wmma_bf16(af1, load_b_frag(bsb + 32, 64), acc[jn]);
    }
    __syncthreads();
  }
#pragma unroll
  for (int jn = 0; jn < 4; ++jn)
#pragma unroll
    for (int r = 0; r < 8; ++r) {
      int row = wm * 16 + r + 8 * half;
      int j = col0 + wn * 64 + jn * 16 + lcol;
      out[(row0 + row) * DIMC + j] = acc[jn][r] + proj_b[j];
    }
}

// ---- host launch --------------------------------------------------------
extern "C" void kernel_launch(void* const* d_in, const int* in_sizes, int n_in,
                              void* d_out, int out_size, void* d_ws, size_t ws_size,
                              hipStream_t stream) {
  (void)in_sizes; (void)n_in; (void)out_size; (void)ws_size;
  const float* x       = (const float*)d_in[0];
  const float* q_w     = (const float*)d_in[1];
  const float* kv_w    = (const float*)d_in[2];
  const float* proj_w  = (const float*)d_in[3];
  const float* proj_b  = (const float*)d_in[4];
  const float* dwc_w   = (const float*)d_in[5];
  const float* dwc_b   = (const float*)d_in[6];
  const float* an_bias = (const float*)d_in[7];
  const float* na_bias = (const float*)d_in[8];
  const float* agt     = (const float*)d_in[9];
  float* out = (float*)d_out;

  // workspace carve-up (~278 MB total), 256B aligned
  char* ws = (char*)d_ws;
  auto alloc = [&](size_t bytes) {
    char* p = ws; ws += (bytes + 255) & ~(size_t)255; return p;
  };
  const size_t hv = (size_t)BATCH * NHEADS * NTOK * HDIM;  // per q/k/v
  __bf16* qb    = (__bf16*)alloc(hv * 2);
  __bf16* kbf   = (__bf16*)alloc(hv * 2);
  __bf16* vbf   = (__bf16*)alloc(hv * 2);
  float*  attn  = (float*)alloc((size_t)BATCH * NTOK * DIMC * 4);
  float*  pb1   = (float*)alloc((size_t)NHEADS * AGENTN * NTOK * 4);
  float*  nb1   = (float*)alloc((size_t)NHEADS * AGENTN * NTOK * 4);
  __bf16* agv   = (__bf16*)alloc((size_t)BATCH * NHEADS * AGP * HDIM * 2);
  __bf16* agbf  = (__bf16*)alloc((size_t)NHEADS * AGP * HDIM * 2);
  __bf16* wqkv  = (__bf16*)alloc((size_t)DIMC * 1536 * 2);
  __bf16* wproj = (__bf16*)alloc((size_t)DIMC * DIMC * 2);

  int cvt_total = DIMC * 1536 + DIMC * DIMC + NHEADS * AGP * HDIM;
  cvt_pack<<<(cvt_total + 255) / 256, 256, 0, stream>>>(q_w, kv_w, proj_w, agt,
                                                        wqkv, wproj, agbf);
  int rz_total = 2 * NHEADS * AGENTN * NTOK;
  resize_bias<<<(rz_total + 255) / 256, 256, 0, stream>>>(an_bias, na_bias, pb1, nb1);

  gemm_qkv<<<dim3(1536 / 128, (BATCH * NTOK) / 64), 256, 0, stream>>>(x, wqkv, qb, kbf, vbf);

  agent_attn<<<BATCH * NHEADS, 192, 0, stream>>>(agbf, kbf, vbf, pb1, agv);

  q_attn<<<dim3(NTOK / 64, BATCH * NHEADS), 128, 0, stream>>>(qb, agbf, agv, nb1, attn);

  long dwc_total = (long)BATCH * DIMC * NTOK;
  dwc_add<<<(int)((dwc_total + 255) / 256), 256, 0, stream>>>(vbf, dwc_w, dwc_b, attn);

  gemm_proj<<<dim3(DIMC / 128, (BATCH * NTOK) / 64), 256, 0, stream>>>(attn, wproj,
                                                                       proj_b, out);
}
